// Net_80796924772493
// MI455X (gfx1250) — matrix operand order repaired and checked by reading, not compiled, not run on hardware
//
#include <hip/hip_runtime.h>
#include <cstdint>
#include <cstddef>

typedef unsigned short u16;
typedef __bf16 bf16_t;
typedef __attribute__((ext_vector_type(16))) __bf16 v16bf;
typedef __attribute__((ext_vector_type(8)))  float  v8f;

__device__ __forceinline__ u16 f2bf(float f) {
  union { float f; unsigned int u; } c; c.f = f;
  unsigned int u = c.u;
  u += 0x7fffu + ((u >> 16) & 1u);          // round-to-nearest-even
  return (u16)(u >> 16);
}

// ---------------------------------------------------------------------------
// Build (c0, k, c1) decode tables for the combinatorial perm matrices.
// Row semantics: g[row] = y[c0] + F[k] - y[c1],  F[k] = sum over filter k's chans
// ---------------------------------------------------------------------------
__global__ void k_build_tables(int* __restrict__ t1, int* __restrict__ t2) {
  if (threadIdx.x != 0 || blockIdx.x != 0) return;
  int cnt = 0;
  for (int i = 0; i < 10; ++i)                    // stage 1: C=30, nf=10, ncpf=3
    for (int j = 0; j < 3; ++j) {
      int ks = (j == 0) ? i : i + 1;
      for (int k = ks; k < 10; ++k) {
        t1[cnt * 3 + 0] = i * 3 + j;
        t1[cnt * 3 + 1] = k;
        t1[cnt * 3 + 2] = k * 3 + j;
        ++cnt;
      }
    }
  cnt = 0;
  for (int i = 0; i < 6; ++i)                     // stage 2: C=600, nf=6, ncpf=100
    for (int j = 0; j < 100; ++j) {
      int ks = (j == 0) ? i : i + 1;
      for (int k = ks; k < 6; ++k) {
        t2[cnt * 3 + 0] = i * 100 + j;
        t2[cnt * 3 + 1] = k;
        t2[cnt * 3 + 2] = k * 100 + j;
        ++cnt;
      }
    }
}

// ---------------------------------------------------------------------------
// f32 -> padded bf16 pack (zero padding outside [SM,SK])
// ---------------------------------------------------------------------------
__global__ void k_pack_bf16(const float* __restrict__ src, u16* __restrict__ dst,
                            int SM, int SK, int DM, int DK) {
  int idx = blockIdx.x * 256 + threadIdx.x;
  if (idx >= DM * DK) return;
  int m = idx / DK, k = idx - m * DK;
  float v = (m < SM && k < SK) ? src[m * SK + k] : 0.0f;
  dst[idx] = f2bf(v);
}

__global__ void k_zero_u16(u16* __restrict__ p, int n) {
  int i = blockIdx.x * 256 + threadIdx.x;
  if (i < n) p[i] = 0;
}

__global__ void k_f32_to_bf16(const float* __restrict__ src, u16* __restrict__ dst, int n) {
  int i = blockIdx.x * 256 + threadIdx.x;
  if (i < n) dst[i] = f2bf(src[i]);
}

// ---------------------------------------------------------------------------
// Depthwise 3x3 convs (pad 1)
// ---------------------------------------------------------------------------
__global__ void k_dwconv1(const float* __restrict__ x,   // NCHW [500,3,32,32]
                          const float* __restrict__ w,   // [30,9]
                          const float* __restrict__ bias,// [30]
                          float* __restrict__ y,         // [30][nimg*1024]
                          int b0, int nimg) {
  int idx = blockIdx.x * 256 + threadIdx.x;
  int P = nimg * 1024;
  if (idx >= 30 * P) return;
  int c = idx / P, rem = idx - c * P;
  int img = rem >> 10, p = rem & 1023;
  int py = p >> 5, px = p & 31;
  int ci = c / 10;                                 // feature_group_count = 3
  const float* src = x + ((size_t)(b0 + img) * 3 + ci) * 1024;
  const float* wt = w + c * 9;
  float acc = bias[c];
  #pragma unroll
  for (int dy = -1; dy <= 1; ++dy) {
    int yy = py + dy; if (yy < 0 || yy > 31) continue;
    #pragma unroll
    for (int dx = -1; dx <= 1; ++dx) {
      int xx = px + dx; if (xx < 0 || xx > 31) continue;
      acc += wt[(dy + 1) * 3 + dx + 1] * src[yy * 32 + xx];
    }
  }
  y[idx] = acc;
}

__global__ void k_dwconv2(const float* __restrict__ in,  // [100][nimg*256]
                          const float* __restrict__ w,   // [600,9]
                          const float* __restrict__ bias,// [600]
                          float* __restrict__ y,         // [600][nimg*256]
                          int nimg) {
  int idx = blockIdx.x * 256 + threadIdx.x;
  int P = nimg * 256;
  if (idx >= 600 * P) return;
  int c = idx / P, rem = idx - c * P;
  int img = rem >> 8, p = rem & 255;
  int py = p >> 4, px = p & 15;
  int g = c / 6;                                   // feature_group_count = 100
  const float* src = in + (size_t)g * P + img * 256;
  const float* wt = w + c * 9;
  float acc = bias[c];
  #pragma unroll
  for (int dy = -1; dy <= 1; ++dy) {
    int yy = py + dy; if (yy < 0 || yy > 15) continue;
    #pragma unroll
    for (int dx = -1; dx <= 1; ++dx) {
      int xx = px + dx; if (xx < 0 || xx > 15) continue;
      acc += wt[(dy + 1) * 3 + dx + 1] * src[yy * 16 + xx];
    }
  }
  y[idx] = acc;
}

// per-filter channel sums: F[f][p] = sum_{c in filter f} y[c][p]
__global__ void k_fsum(const float* __restrict__ y, float* __restrict__ F,
                       int cpf, int nf, int P) {
  int idx = blockIdx.x * 256 + threadIdx.x;
  if (idx >= nf * P) return;
  int f = idx / P, p = idx - f * P;
  const float* base = y + (size_t)f * cpf * P + p;
  float s = 0.0f;
  for (int c = 0; c < cpf; ++c) s += base[(size_t)c * P];
  F[idx] = s;
}

// g[row] = relu(y[c0] + F[k] - y[c1]) -> bf16, rows [G, Gp) zero-padded
__global__ void k_gbuild(const float* __restrict__ y, const float* __restrict__ F,
                         const int* __restrict__ tab, u16* __restrict__ g,
                         int G, int Gp, int P) {
  int idx = blockIdx.x * 256 + threadIdx.x;
  if (idx >= Gp * P) return;
  int r = idx / P, p = idx - r * P;
  if (r >= G) { g[idx] = 0; return; }
  int c0 = tab[r * 3 + 0], k = tab[r * 3 + 1], c1 = tab[r * 3 + 2];
  float v = y[(size_t)c0 * P + p] + F[(size_t)k * P + p] - y[(size_t)c1 * P + p];
  g[idx] = f2bf(v > 0.0f ? v : 0.0f);
}

// ---------------------------------------------------------------------------
// bf16 WMMA GEMM:  C[Mp][N] f32 = A[Mp][K] bf16 * B[K][N] bf16  (+bias, relu)
// block tile 64(M) x 256(N), K-step 32; 8 waves (2x4), each wave 32x64 ->
// 8 independent v_wmma_f32_16x16x32_bf16 per K-step (A frags reused 4x, B 2x).
// ---------------------------------------------------------------------------
#define LDT 40   // LDS row stride in u16 (padding -> conflict-free ds reads)

__global__ __launch_bounds__(256)
void k_gemm_bf16(const u16* __restrict__ A, const u16* __restrict__ B,
                 float* __restrict__ C, const float* __restrict__ bias,
                 int N, int K, int bias_len, int relu) {
  __shared__ __align__(16) u16 As[64 * LDT];    // A tile 64(M) x 32(K), row-major
  __shared__ __align__(16) u16 Bs[256 * LDT];   // B tile 256(N) x 32(K), transposed
  const int t = threadIdx.x;
  const int wave = t >> 5, lane = t & 31;
  const int wm = wave & 1, wn = wave >> 1;      // 2 x 4 wave grid
  const int m0 = blockIdx.y << 6;
  const int n0 = blockIdx.x << 8;
  const int ln = lane & 15, lh = lane >> 4;
  const int am = t >> 2, akg = (t & 3) << 3;    // A fill: 64 rows x 4 oct-chunks

  v8f acc[2][4] = {};

  for (int k0 = 0; k0 < K; k0 += 32) {
    __syncthreads();
    // stage A tile (16B per thread)
    *(uint4*)&As[am * LDT + akg] =
        *(const uint4*)(A + (size_t)(m0 + am) * K + (k0 + akg));
    // stage B tile: 2 tasks/thread; each loads a 4(K)x4(N) block, transposes in
    // registers, and writes 4 aligned 8B LDS stores into Bs[n][k]
    #pragma unroll
    for (int it = 0; it < 2; ++it) {
      int q = t + (it << 8);
      int bkq = q >> 6, bng = q & 63;           // 8 k-quads x 64 n-quads
      u16 tmp[4][4];
      #pragma unroll
      for (int r = 0; r < 4; ++r) {
        ushort4 v = *(const ushort4*)(B + (size_t)(k0 + bkq * 4 + r) * N + (n0 + bng * 4));
        tmp[r][0] = v.x; tmp[r][1] = v.y; tmp[r][2] = v.z; tmp[r][3] = v.w;
      }
      #pragma unroll
      for (int u = 0; u < 4; ++u) {
        ushort4 w;
        w.x = tmp[0][u]; w.y = tmp[1][u]; w.z = tmp[2][u]; w.w = tmp[3][u];
        *(ushort4*)&Bs[(bng * 4 + u) * LDT + bkq * 4] = w;
      }
    }
    if (k0 + 32 < K)   // hint next B tile toward the caches
      __builtin_prefetch(B + (size_t)(k0 + 32 + (t & 31)) * N + n0 + ((t >> 5) << 5), 0, 1);
    __syncthreads();

    // fragments per documented 16-bit A(16x32) / B(32x16) lane layouts
    v16bf a0, a1, bf0, bf1, bf2, bf3;
    const bf16_t* Ar0 = (const bf16_t*)&As[(wm * 32 + ln) * LDT];
    const bf16_t* Ar1 = (const bf16_t*)&As[(wm * 32 + 16 + ln) * LDT];
    const bf16_t* Br0 = (const bf16_t*)&Bs[(wn * 64 + ln) * LDT];
    const bf16_t* Br1 = (const bf16_t*)&Bs[(wn * 64 + 16 + ln) * LDT];
    const bf16_t* Br2 = (const bf16_t*)&Bs[(wn * 64 + 32 + ln) * LDT];
    const bf16_t* Br3 = (const bf16_t*)&Bs[(wn * 64 + 48 + ln) * LDT];
    #pragma unroll
    for (int e = 0; e < 16; ++e) {
      int ka = (e & 7) + ((e >> 3) << 4) + (lh << 3);
      int kb = (e & 7) + ((e >> 3) << 4) + (lh << 3);
      a0[e] = Ar0[ka];   a1[e] = Ar1[ka];
      bf0[e] = Br0[kb];  bf1[e] = Br1[kb];
      bf2[e] = Br2[kb];  bf3[e] = Br3[kb];
    }
    acc[0][0] = __builtin_amdgcn_wmma_f32_16x16x32_bf16(false, a0, false, bf0, (short)0, acc[0][0], false, false);
    acc[0][1] = __builtin_amdgcn_wmma_f32_16x16x32_bf16(false, a0, false, bf1, (short)0, acc[0][1], false, false);
    acc[0][2] = __builtin_amdgcn_wmma_f32_16x16x32_bf16(false, a0, false, bf2, (short)0, acc[0][2], false, false);
    acc[0][3] = __builtin_amdgcn_wmma_f32_16x16x32_bf16(false, a0, false, bf3, (short)0, acc[0][3], false, false);
    acc[1][0] = __builtin_amdgcn_wmma_f32_16x16x32_bf16(false, a1, false, bf0, (short)0, acc[1][0], false, false);
    acc[1][1] = __builtin_amdgcn_wmma_f32_16x16x32_bf16(false, a1, false, bf1, (short)0, acc[1][1], false, false);
    acc[1][2] = __builtin_amdgcn_wmma_f32_16x16x32_bf16(false, a1, false, bf2, (short)0, acc[1][2], false, false);
    acc[1][3] = __builtin_amdgcn_wmma_f32_16x16x32_bf16(false, a1, false, bf3, (short)0, acc[1][3], false, false);
  }

  // epilogue: bias + optional relu, f32 store
  #pragma unroll
  for (int sm = 0; sm < 2; ++sm) {
    #pragma unroll
    for (int sn = 0; sn < 4; ++sn) {
      #pragma unroll
      for (int vi = 0; vi < 8; ++vi) {
        int m = m0 + wm * 32 + sm * 16 + vi + (lh << 3);
        int n = n0 + wn * 64 + sn * 16 + ln;
        float bv = (bias != nullptr && m < bias_len) ? bias[m] : 0.0f;
        float val = acc[sm][sn][vi] + bv;
        if (relu) val = val > 0.0f ? val : 0.0f;
        C[(size_t)m * N + n] = val;
      }
    }
  }
}

// ---------------------------------------------------------------------------
// 2x2 max pools
// ---------------------------------------------------------------------------
__global__ void k_pool1(const float* __restrict__ in,  // [>=C][nimg*32*32] (already relu'd)
                        float* __restrict__ out,       // [C][nimg*16*16]
                        int C, int nimg) {
  int Pin = nimg * 1024, Po = nimg * 256;
  int idx = blockIdx.x * 256 + threadIdx.x;
  if (idx >= C * Po) return;
  int c = idx / Po, rem = idx - c * Po;
  int img = rem >> 8, p = rem & 255;
  int oy = p >> 4, ox = p & 15;
  const float* s = in + (size_t)c * Pin + img * 1024;
  int iy = oy << 1, ix = ox << 1;
  float m = s[iy * 32 + ix];
  m = fmaxf(m, s[iy * 32 + ix + 1]);
  m = fmaxf(m, s[(iy + 1) * 32 + ix]);
  m = fmaxf(m, s[(iy + 1) * 32 + ix + 1]);
  out[idx] = m;
}

// pool stage2 output straight into the transposed bf16 fc1 input [19200][512]
__global__ void k_pool2_fc(const float* __restrict__ in,  // [>=300][nimg*256]
                           u16* __restrict__ fcinT, int b0, int nimg) {
  int P2 = nimg * 256;
  int idx = blockIdx.x * 256 + threadIdx.x;
  if (idx >= 300 * nimg * 64) return;
  int c = idx / (nimg * 64), rem = idx - c * (nimg * 64);
  int img = rem >> 6, p = rem & 63;
  int oy = p >> 3, ox = p & 7;
  const float* s = in + (size_t)c * P2 + img * 256;
  int iy = oy << 1, ix = ox << 1;
  float m = s[iy * 16 + ix];
  m = fmaxf(m, s[iy * 16 + ix + 1]);
  m = fmaxf(m, s[(iy + 1) * 16 + ix]);
  m = fmaxf(m, s[(iy + 1) * 16 + ix + 1]);
  fcinT[(size_t)(c * 64 + p) * 512 + (b0 + img)] = f2bf(m);
}

__global__ void k_final(const float* __restrict__ fc2out, float* __restrict__ out) {
  int idx = blockIdx.x * 256 + threadIdx.x;
  if (idx >= 5000) return;
  int b = idx / 10, o = idx - b * 10;
  out[idx] = fc2out[(size_t)o * 512 + b];
}

// ---------------------------------------------------------------------------
// Host launch
// ---------------------------------------------------------------------------
#define GRID1(n) dim3(((n) + 255) / 256)

extern "C" void kernel_launch(void* const* d_in, const int* in_sizes, int n_in,
                              void* d_out, int out_size, void* d_ws, size_t ws_size,
                              hipStream_t stream) {
  (void)in_sizes; (void)n_in; (void)out_size; (void)ws_size;

  const float* x     = (const float*)d_in[0];
  const float* dw1_w = (const float*)d_in[1];
  const float* dw1_b = (const float*)d_in[2];
  const float* pw1_w = (const float*)d_in[3];
  const float* pw1_b = (const float*)d_in[4];
  const float* dw2_w = (const float*)d_in[5];
  const float* dw2_b = (const float*)d_in[6];
  const float* pw2_w = (const float*)d_in[7];
  const float* pw2_b = (const float*)d_in[8];
  const float* fc1_w = (const float*)d_in[9];
  const float* fc1_b = (const float*)d_in[10];
  const float* fc2_w = (const float*)d_in[11];
  const float* fc2_b = (const float*)d_in[12];
  float* out = (float*)d_out;

  const int CHUNK = 50;              // 10 chunks of 50 images
  const int P1 = CHUNK * 1024;       // 51200 stage-1 pixels / chunk
  const int P2 = CHUNK * 256;        // 12800 stage-2 pixels / chunk

  // workspace carve-up (256B aligned)
  size_t off = 0;
  auto give = [&](size_t bytes) -> void* {
    void* p = (char*)d_ws + off;
    off += (bytes + 255) & ~(size_t)255;
    return p;
  };
  int*   tab1   = (int*)  give(145  * 3 * sizeof(int));
  int*   tab2   = (int*)  give(1506 * 3 * sizeof(int));
  u16*   Apw1   = (u16*)  give((size_t)128 * 160   * 2);   // pw1_w padded
  u16*   Apw2   = (u16*)  give((size_t)320 * 1536  * 2);   // pw2_w padded
  u16*   Afc1   = (u16*)  give((size_t)128 * 19200 * 2);   // fc1_w padded
  u16*   Afc2   = (u16*)  give((size_t)64  * 128   * 2);   // fc2_w padded (M 10->64)
  u16*   fcinT  = (u16*)  give((size_t)19200 * 512 * 2);   // fc1 input, transposed
  float* fc1out = (float*)give((size_t)128 * 512 * 4);
  u16*   h1bf   = (u16*)  give((size_t)128 * 512 * 2);
  float* fc2out = (float*)give((size_t)64  * 512 * 4);
  float* y1     = (float*)give((size_t)30   * P1 * 4);
  float* F1     = (float*)give((size_t)10   * P1 * 4);
  u16*   g1     = (u16*)  give((size_t)160  * P1 * 2);
  float* out1   = (float*)give((size_t)128  * P1 * 4);
  float* pool1  = (float*)give((size_t)100  * P2 * 4);
  float* y2     = (float*)give((size_t)600  * P2 * 4);
  float* F2     = (float*)give((size_t)6    * P2 * 4);
  u16*   g2     = (u16*)  give((size_t)1536 * P2 * 2);
  float* out2   = (float*)give((size_t)320  * P2 * 4);

  // one-time (per launch) setup
  k_build_tables<<<1, 1, 0, stream>>>(tab1, tab2);
  k_pack_bf16<<<GRID1(128 * 160),   256, 0, stream>>>(pw1_w, Apw1, 100, 145,  128, 160);
  k_pack_bf16<<<GRID1(320 * 1536),  256, 0, stream>>>(pw2_w, Apw2, 300, 1506, 320, 1536);
  k_pack_bf16<<<GRID1(128 * 19200), 256, 0, stream>>>(fc1_w, Afc1, 120, 19200, 128, 19200);
  k_pack_bf16<<<GRID1(64 * 128),    256, 0, stream>>>(fc2_w, Afc2, 10, 120,   64, 128);
  k_zero_u16<<<GRID1(19200 * 512),  256, 0, stream>>>(fcinT, 19200 * 512);

  for (int c = 0; c < 500 / CHUNK; ++c) {
    int b0 = c * CHUNK;
    // stage 1
    k_dwconv1<<<GRID1(30 * P1), 256, 0, stream>>>(x, dw1_w, dw1_b, y1, b0, CHUNK);
    k_fsum   <<<GRID1(10 * P1), 256, 0, stream>>>(y1, F1, 3, 10, P1);
    k_gbuild <<<GRID1(160 * P1), 256, 0, stream>>>(y1, F1, tab1, g1, 145, 160, P1);
    k_gemm_bf16<<<dim3(P1 / 256, 128 / 64), 256, 0, stream>>>(Apw1, g1, out1, pw1_b,
                                                              P1, 160, 100, 1);
    k_pool1  <<<GRID1(100 * P2), 256, 0, stream>>>(out1, pool1, 100, CHUNK);
    // stage 2
    k_dwconv2<<<GRID1(600 * P2), 256, 0, stream>>>(pool1, dw2_w, dw2_b, y2, CHUNK);
    k_fsum   <<<GRID1(6 * P2), 256, 0, stream>>>(y2, F2, 100, 6, P2);
    k_gbuild <<<GRID1(1536 * P2), 256, 0, stream>>>(y2, F2, tab2, g2, 1506, 1536, P2);
    k_gemm_bf16<<<dim3(P2 / 256, 320 / 64), 256, 0, stream>>>(Apw2, g2, out2, pw2_b,
                                                              P2, 1536, 300, 1);
    k_pool2_fc<<<GRID1(300 * CHUNK * 64), 256, 0, stream>>>(out2, fcinT, b0, CHUNK);
  }

  // fc1: [128(pad120),19200] x [19200,512(pad500)] -> [128][512], no relu
  k_gemm_bf16<<<dim3(512 / 256, 128 / 64), 256, 0, stream>>>(Afc1, fcinT, fc1out, fc1_b,
                                                             512, 19200, 120, 0);
  k_f32_to_bf16<<<GRID1(128 * 512), 256, 0, stream>>>(fc1out, h1bf, 128 * 512);
  // fc2: [64(pad10),128] x [128,512] -> [64][512]
  k_gemm_bf16<<<dim3(512 / 256, 64 / 64), 256, 0, stream>>>(Afc2, h1bf, fc2out, fc2_b,
                                                            512, 128, 10, 0);
  k_final<<<GRID1(5000), 256, 0, stream>>>(fc2out, out);
}
